// MultiCellLSTM_82154134438524
// MI455X (gfx1250) — matrix operand order repaired and checked by reading, not compiled, not run on hardware
//
#include <hip/hip_runtime.h>

typedef __attribute__((ext_vector_type(16))) _Float16 v16h;
typedef __attribute__((ext_vector_type(8)))  float    v8f;
typedef __attribute__((ext_vector_type(4)))  unsigned u32x4;
typedef __attribute__((ext_vector_type(2)))  unsigned u32x2;
typedef __attribute__((ext_vector_type(4)))  float    f32x4;

#define HDIM    64
#define FOURH   256
#define T1C     512
#define T2C     256
#define T3C     128
#define BROWS   16      // batch rows per workgroup
#define NTHR    128     // 4 waves

// Padded LDS strides (non-power-of-two, 16B-aligned) to avoid bank conflicts
#define WH_STR   72     // halves per Wh row (64 used + 8 pad) = 144 B
#define WH_STRU  36     // u32 per Wh row
#define HB_STR   72     // halves per hbuf row
#define GB_STR   20     // floats per gbuf column (16 used + 4 pad) = 80 B

union Frag { v16h v; u32x4 u[2]; };
union Acc  { v8f  v; f32x4 f[2]; };

__device__ __forceinline__ unsigned pkh(float a, float b) {
  union { _Float16 h[2]; unsigned u; } z;
  z.h[0] = (_Float16)a; z.h[1] = (_Float16)b;
  return z.u;
}
// Branch-free activations: v_exp_f32 + v_rcp_f32, no OCML library path.
__device__ __forceinline__ float sigf(float x) {
  return __builtin_amdgcn_rcpf(1.0f + __expf(-x));
}
__device__ __forceinline__ float tanh_fast(float x) {
  // 1 - 2/(1+e^{2x}); exact at +-inf (rcp(inf)=0), abs err ~2e-7 near 0
  return 1.0f - 2.0f * __builtin_amdgcn_rcpf(1.0f + __expf(2.0f * x));
}

__global__ __launch_bounds__(NTHR)
void MultiCellLSTM_kernel(
    const float* __restrict__ x1, const float* __restrict__ x2, const float* __restrict__ x3,
    const float* __restrict__ Wi0, const float* __restrict__ Wh0, const float* __restrict__ bi0, const float* __restrict__ bh0,
    const float* __restrict__ Wi1, const float* __restrict__ Wh1, const float* __restrict__ bi1, const float* __restrict__ bh1,
    const float* __restrict__ Wi2, const float* __restrict__ Wh2, const float* __restrict__ bi2, const float* __restrict__ bh2,
    const float* __restrict__ Wout, const float* __restrict__ bout,
    float* __restrict__ out)
{
  // LDS-resident state & weights (~140 KB of the 320 KB/WGP -> 2 WGs/WGP)
  __shared__ __align__(16) _Float16 whH[3 * FOURH * WH_STR];  // 108 KB [cell][row][k(padded)]
  __shared__ __align__(16) u32x2    wixH[3 * FOURH];          //   6 KB per row: pack(wi0,wi1), pack(wi2,bias)
  __shared__ __align__(16) float    gbuf[FOURH * GB_STR];     //  20 KB [gate_col][m(padded)]
  __shared__ __align__(16) _Float16 hbuf[BROWS * HB_STR];     // 2.3 KB [m][k(padded)]
  __shared__               float    outacc[BROWS];

  const int tid    = threadIdx.x;
  const int lane   = tid & 31;
  const int wave   = tid >> 5;
  const int wgBase = blockIdx.x * BROWS;

  // ---------------- preload: convert weights to f16 in LDS ----------------
  {
    const float* WhPtr[3] = {Wh0, Wh1, Wh2};
    unsigned* whU = (unsigned*)whH;
    for (int c = 0; c < 3; ++c) {
      for (int r = tid; r < FOURH; r += NTHR) {
        const float* W = WhPtr[c] + r * HDIM;
        unsigned* dst  = whU + (c * FOURH + r) * WH_STRU;
        #pragma unroll
        for (int k = 0; k < HDIM / 2; ++k)
          dst[k] = pkh(W[2 * k], W[2 * k + 1]);
      }
    }
    const float* WiPtr[3] = {Wi0, Wi1, Wi2};
    const float* biPtr[3] = {bi0, bi1, bi2};
    const float* bhPtr[3] = {bh0, bh1, bh2};
    const int    wW[3]    = {3, 2, 1};
    for (int c = 0; c < 3; ++c) {
      int w = wW[c];
      for (int r = tid; r < FOURH; r += NTHR) {
        float a  = WiPtr[c][r * w];
        float b  = (w > 1) ? WiPtr[c][r * w + 1] : 0.0f;
        float d  = (w > 2) ? WiPtr[c][r * w + 2] : 0.0f;
        float bs = biPtr[c][r] + bhPtr[c][r];
        u32x2 p; p.x = pkh(a, b); p.y = pkh(d, bs);
        wixH[c * FOURH + r] = p;
      }
    }
    unsigned* hU = (unsigned*)hbuf;
    for (int i = tid; i < (BROWS * HB_STR) / 2; i += NTHR) hU[i] = 0u;
    if (tid < BROWS) outacc[tid] = 0.0f;
  }
  __syncthreads();

  float cst[8];
  #pragma unroll
  for (int k = 0; k < 8; ++k) cst[k] = 0.0f;

  const int m16    = lane & 15;
  const int hiHalf = (lane < 16) ? 0 : 1;   // lane-group selector
  const int mstore = hiHalf * 8;            // C/D rows covered by this lane

  for (int t = 0; t < T1C; ++t) {
    const int ty = ((t & 3) == 0) ? 0 : (((t & 1) == 0) ? 1 : 2);

    // ---- A fragments (h, f16) from hbuf: 16x32 per fragment ----
    Frag a0, a1, ax;
    {
      const _Float16* hrow = hbuf + m16 * HB_STR;
      const int kb0 = hiHalf * 8;        // lanes<16: K 0..7 & 16..23 ; lanes>=16: 8..15 & 24..31
      a0.u[0] = *(const u32x4*)(hrow + kb0);
      a0.u[1] = *(const u32x4*)(hrow + kb0 + 16);
      const int kb1 = 32 + hiHalf * 8;
      a1.u[0] = *(const u32x4*)(hrow + kb1);
      a1.u[1] = *(const u32x4*)(hrow + kb1 + 16);
    }
    // ---- A_x: [x1, x2|0, x3|0, 1] in K=0..3, zeros elsewhere ----
    {
      u32x4 z = {0u, 0u, 0u, 0u};
      ax.u[0] = z; ax.u[1] = z;
    }
    if (lane < 16) {
      const int row = wgBase + lane;
      float xv1 = x1[row * T1C + t];
      float xv2 = ((t & 1) == 0) ? x2[row * T2C + (t >> 1)] : 0.0f;
      float xv3 = ((t & 3) == 0) ? x3[row * T3C + (t >> 2)] : 0.0f;
      ax.u[0].x = pkh(xv1, xv2);
      ax.u[0].y = pkh(xv3, 1.0f);
    }

    const _Float16* wbase  = whH + (ty * FOURH) * WH_STR;
    const u32x2*    wxbase = wixH + ty * FOURH;

    #pragma unroll
    for (int tt = 0; tt < 4; ++tt) {
      const int col = wave * 64 + tt * 16 + m16;   // absolute gate column 0..255

      // B_x: Wi columns + bias in K=0..3 (lanes<16 hold K 0..7 & 16..23)
      Frag bx, b0, b1;
      { u32x4 z = {0u,0u,0u,0u}; bx.u[0] = z; bx.u[1] = z; }
      { u32x2 wx = wxbase[col]; bx.u[0].x = wx.x; bx.u[0].y = wx.y; }

      // B_w: Wh^T tiles; lane<16 -> K 0..7 & 16..23, lane>=16 -> K 8..15 & 24..31 (per K-half)
      const _Float16* wrow = wbase + col * WH_STR;
      const int kbB0 = hiHalf * 8;
      b0.u[0] = *(const u32x4*)(wrow + kbB0);
      b0.u[1] = *(const u32x4*)(wrow + kbB0 + 16);
      const int kbB1 = 32 + hiHalf * 8;
      b1.u[0] = *(const u32x4*)(wrow + kbB1);
      b1.u[1] = *(const u32x4*)(wrow + kbB1 + 16);

      Acc acc;
      #pragma unroll
      for (int i = 0; i < 8; ++i) acc.v[i] = 0.0f;
      acc.v = __builtin_amdgcn_wmma_f32_16x16x32_f16(false, ax.v, false, bx.v, (short)0, acc.v, false, false);
      acc.v = __builtin_amdgcn_wmma_f32_16x16x32_f16(false, a0.v, false, b0.v, (short)0, acc.v, false, false);
      acc.v = __builtin_amdgcn_wmma_f32_16x16x32_f16(false, a1.v, false, b1.v, (short)0, acc.v, false, false);

      // store pre-activations: acc[v] -> (m = mstore+v, col); gbuf is [col][m]
      float* gp = gbuf + col * GB_STR + mstore;
      *(f32x4*)(gp)     = acc.f[0];
      *(f32x4*)(gp + 4) = acc.f[1];
    }
    __syncthreads();

    // ---- elementwise LSTM update; c lives in registers ----
    {
      const int em = tid & 15;            // batch row in tile
      const int j0 = (tid >> 4) << 3;     // 8 hidden units per thread
      float hv[8];
      #pragma unroll
      for (int k = 0; k < 8; ++k) {
        const int j = j0 + k;
        float ig = gbuf[(j)        * GB_STR + em];
        float fg = gbuf[(64  + j)  * GB_STR + em];
        float gg = gbuf[(128 + j)  * GB_STR + em];
        float og = gbuf[(192 + j)  * GB_STR + em];
        float cc = sigf(fg) * cst[k] + sigf(ig) * tanh_fast(gg);
        cst[k] = cc;
        hv[k]  = sigf(og) * tanh_fast(cc);
      }
      u32x4 hp;
      hp.x = pkh(hv[0], hv[1]); hp.y = pkh(hv[2], hv[3]);
      hp.z = pkh(hv[4], hv[5]); hp.w = pkh(hv[6], hv[7]);
      *(u32x4*)(hbuf + em * HB_STR + j0) = hp;

      if (t == T1C - 1) {                  // fold final h @ W_out^T
        float p = 0.0f;
        #pragma unroll
        for (int k = 0; k < 8; ++k) p += hv[k] * Wout[j0 + k];
        atomicAdd(&outacc[em], p);
      }
    }
    __syncthreads();
  }

  if (tid < BROWS) {
    float v = outacc[tid] + bout[0];
    out[wgBase + tid] = sigf(v);
  }
}

extern "C" void kernel_launch(void* const* d_in, const int* in_sizes, int n_in,
                              void* d_out, int out_size, void* d_ws, size_t ws_size,
                              hipStream_t stream) {
  (void)n_in; (void)out_size; (void)d_ws; (void)ws_size;
  const float* x1   = (const float*)d_in[0];
  const float* x2   = (const float*)d_in[1];
  const float* x3   = (const float*)d_in[2];
  const float* Wi0  = (const float*)d_in[3];   // W_ih3 [256,3] -> step type 0
  const float* Wh0  = (const float*)d_in[4];
  const float* bi0  = (const float*)d_in[5];
  const float* bh0  = (const float*)d_in[6];
  const float* Wi1  = (const float*)d_in[7];   // W_ih2 [256,2] -> type 1
  const float* Wh1  = (const float*)d_in[8];
  const float* bi1  = (const float*)d_in[9];
  const float* bh1  = (const float*)d_in[10];
  const float* Wi2  = (const float*)d_in[11];  // W_ih1 [256,1] -> type 2
  const float* Wh2  = (const float*)d_in[12];
  const float* bi2  = (const float*)d_in[13];
  const float* bh2  = (const float*)d_in[14];
  const float* Wout = (const float*)d_in[15];
  const float* bout = (const float*)d_in[16];
  float* out = (float*)d_out;

  const int B    = in_sizes[0] / T1C;  // 4096
  const int grid = B / BROWS;          // 256 workgroups x 128 threads
  MultiCellLSTM_kernel<<<grid, NTHR, 0, stream>>>(
      x1, x2, x3,
      Wi0, Wh0, bi0, bh0,
      Wi1, Wh1, bi1, bh1,
      Wi2, Wh2, bi2, bh2,
      Wout, bout, out);
}
